// GCNEncoder_30562987278563
// MI455X (gfx1250) — compile-verified, matrix operand order checked by reading, not run on hardware
//
#include <hip/hip_runtime.h>
#include <cstdint>
#include <cstddef>

// ---------------------------------------------------------------------------
// 2-layer GCN for MI455X (gfx1250, wave32).
//   norm[e] = dinv[row]*dinv[col];  h = relu(S (x@W1) + b1);  out = S (h@W2) + b2
// Dense transforms via v_wmma_f32_16x16x32_bf16; edge phase via coalesced
// float4 gathers + global_atomic_add_f32 scatter (memory-bound part).
// ---------------------------------------------------------------------------

#define IN_C 128
#define HIDC 128
#define OUTC 64

typedef __attribute__((ext_vector_type(16))) __bf16 v16bf;
typedef __attribute__((ext_vector_type(8)))  float  v8f;

__device__ __forceinline__ void atomAddF(float* p, float v) {
  __hip_atomic_fetch_add(p, v, __ATOMIC_RELAXED, __HIP_MEMORY_SCOPE_AGENT);
}

// ---------------- degree / norm ----------------
__global__ void gcn_deg_kernel(const int* __restrict__ col, float* __restrict__ deg, int E) {
  int e = blockIdx.x * blockDim.x + threadIdx.x;
  if (e < E) atomAddF(&deg[col[e]], 1.0f);
}

__global__ void gcn_dinv_kernel(float* __restrict__ deg, int N) {
  int i = blockIdx.x * blockDim.x + threadIdx.x;
  if (i < N) {
    float d = deg[i];
    deg[i] = (d > 0.0f) ? rsqrtf(d) : 0.0f;
  }
}

__global__ void gcn_norm_kernel(const int* __restrict__ row, const int* __restrict__ col,
                                const float* __restrict__ dinv, float* __restrict__ norm, int E) {
  int e = blockIdx.x * blockDim.x + threadIdx.x;
  if (e < E) norm[e] = dinv[row[e]] * dinv[col[e]];
}

// ---------------- weight convert: W[KxN] f32 -> Wt[NxK] bf16 ----------------
__global__ void gcn_convert_w(const float* __restrict__ W, __bf16* __restrict__ Wt,
                              int K, int Ncol) {
  int idx = blockIdx.x * blockDim.x + threadIdx.x;
  if (idx >= K * Ncol) return;
  int n = idx / K;
  int k = idx - n * K;
  Wt[idx] = (__bf16)W[k * Ncol + n];
}

// ---------------- WMMA GEMM: C[MxNCOL] = A[Mx128](f32) * W(bf16, Wt=[NCOLx128]) ----
// One wave computes a 16-row stripe across all NCOL columns (A fragment reused
// NCOL/16 times). W is staged in LDS; B fragments come from ds_load.
template <int NCOL>
__global__ void __launch_bounds__(256)
gcn_wmma_gemm(const float* __restrict__ A, const __bf16* __restrict__ Wt,
              float* __restrict__ C, int M) {
  constexpr int K  = 128;
  constexpr int NT = NCOL / 16;

  __shared__ __bf16 sW[NCOL * K];
  {
    const uint4* src = reinterpret_cast<const uint4*>(Wt);
    uint4*       dst = reinterpret_cast<uint4*>(sW);
    constexpr int TOT = NCOL * K * 2 / 16;  // uint4 chunks
    for (int i = threadIdx.x; i < TOT; i += 256) dst[i] = src[i];
  }
  __syncthreads();

  const int waveId = blockIdx.x * 8 + (threadIdx.x >> 5);
  const int mBase  = waveId * 16;
  if (mBase >= M) return;

  const int lane = threadIdx.x & 31;
  const int half = lane >> 4;   // selects K-group per ISA 16-bit layout
  const int ml   = lane & 15;   // M (for A) / N (for B,C) within tile

  v8f acc[NT];
#pragma unroll
  for (int t = 0; t < NT; ++t)
#pragma unroll
    for (int j = 0; j < 8; ++j) acc[t][j] = 0.0f;

  const float* arow = A + (size_t)(mBase + ml) * K;
  __builtin_prefetch(arow, 0, 1);  // global_prefetch_b8

#pragma unroll
  for (int kb = 0; kb < K; kb += 32) {
    // A fragment 16x32 bf16: VGPR i<4 -> K = kb + half*8 + 2i ; i>=4 -> +16
    v16bf a;
#pragma unroll
    for (int i = 0; i < 4; ++i) {
      float2 lo = *reinterpret_cast<const float2*>(arow + kb + half * 8 + 2 * i);
      float2 hi = *reinterpret_cast<const float2*>(arow + kb + 16 + half * 8 + 2 * i);
      a[2 * i]     = (__bf16)lo.x;
      a[2 * i + 1] = (__bf16)lo.y;
      a[8 + 2 * i] = (__bf16)hi.x;
      a[9 + 2 * i] = (__bf16)hi.y;
    }
#pragma unroll
    for (int t = 0; t < NT; ++t) {
      const __bf16* br = sW + (size_t)(t * 16 + ml) * K;  // row n of Wt = column n of W
      v16bf b;
#pragma unroll
      for (int i = 0; i < 4; ++i) {
        b[2 * i]     = br[kb + half * 8 + 2 * i];
        b[2 * i + 1] = br[kb + half * 8 + 2 * i + 1];
        b[8 + 2 * i] = br[kb + 16 + half * 8 + 2 * i];
        b[9 + 2 * i] = br[kb + 16 + half * 8 + 2 * i + 1];
      }
      acc[t] = __builtin_amdgcn_wmma_f32_16x16x32_bf16(
          false, a, false, b, (short)0, acc[t], false, false);
    }
  }

  // f32 C/D layout: lane n = ml, VGPR j -> row mBase + half*8 + j
#pragma unroll
  for (int t = 0; t < NT; ++t) {
#pragma unroll
    for (int j = 0; j < 8; ++j) {
      C[(size_t)(mBase + half * 8 + j) * NCOL + t * 16 + ml] = acc[t][j];
    }
  }
}

// ---------------- bias broadcast init ----------------
template <int C>
__global__ void gcn_init_bias(float* __restrict__ dst, const float* __restrict__ b,
                              long long total) {
  long long i = (long long)blockIdx.x * blockDim.x + threadIdx.x;
  if (i < total) dst[i] = b[(int)(i & (C - 1))];
}

// ---------------- relu ----------------
__global__ void gcn_relu(float* __restrict__ x, long long total) {
  long long i = (long long)blockIdx.x * blockDim.x + threadIdx.x;
  if (i < total) x[i] = fmaxf(x[i], 0.0f);
}

// ---------------- edge scatter: dst[col] += src[row] * norm ----------------
template <int C>
__global__ void gcn_scatter(const float* __restrict__ src, const float* __restrict__ norm,
                            const int* __restrict__ row, const int* __restrict__ col,
                            float* __restrict__ dst, int E) {
  constexpr int TPE = C / 4;  // threads per edge (float4 each)
  long long tid = (long long)blockIdx.x * blockDim.x + threadIdx.x;
  long long e   = tid / TPE;
  if (e >= E) return;
  int f = ((int)tid & (TPE - 1)) * 4;
  int r = row[e];
  int c = col[e];
  float w = norm[e];
  const float4 v = *reinterpret_cast<const float4*>(src + (size_t)r * C + f);
  float* d = dst + (size_t)c * C + f;
  atomAddF(d + 0, v.x * w);
  atomAddF(d + 1, v.y * w);
  atomAddF(d + 2, v.z * w);
  atomAddF(d + 3, v.w * w);
}

// ---------------------------------------------------------------------------
extern "C" void kernel_launch(void* const* d_in, const int* in_sizes, int n_in,
                              void* d_out, int out_size, void* d_ws, size_t ws_size,
                              hipStream_t stream) {
  const float* x  = (const float*)d_in[0];
  const int*   ei = (const int*)d_in[1];
  const float* W1 = (const float*)d_in[2];
  const float* b1 = (const float*)d_in[3];
  const float* W2 = (const float*)d_in[4];
  const float* b2 = (const float*)d_in[5];

  const int N = in_sizes[0] / IN_C;
  const int E = in_sizes[1] / 2;
  const int* row = ei;        // edge_index[0] = source
  const int* col = ei + E;    // edge_index[1] = target

  // --- workspace carve-up (16B aligned regions) ---
  char* ws = (char*)d_ws;
  float* deg = (float*)ws;           ws += (size_t)N * 4;        // reused as dinv
  float* nrm = (float*)ws;           ws += (size_t)E * 4;
  __bf16* W1t = (__bf16*)ws;         ws += (size_t)HIDC * IN_C * 2;
  __bf16* W2t = (__bf16*)ws;         ws += (size_t)OUTC * HIDC * 2;
  float* h1  = (float*)ws;           ws += (size_t)N * HIDC * 4; // reused as h2
  float* agg1 = (float*)ws;          ws += (size_t)N * HIDC * 4;
  float* h2 = h1;
  float* out = (float*)d_out;

  // 1) normalization coefficients
  hipMemsetAsync(deg, 0, (size_t)N * 4, stream);
  gcn_deg_kernel<<<(E + 255) / 256, 256, 0, stream>>>(col, deg, E);
  gcn_dinv_kernel<<<(N + 255) / 256, 256, 0, stream>>>(deg, N);
  gcn_norm_kernel<<<(E + 255) / 256, 256, 0, stream>>>(row, col, deg, nrm, E);

  // 2) weights -> transposed bf16
  gcn_convert_w<<<(IN_C * HIDC + 255) / 256, 256, 0, stream>>>(W1, W1t, IN_C, HIDC);
  gcn_convert_w<<<(HIDC * OUTC + 255) / 256, 256, 0, stream>>>(W2, W2t, HIDC, OUTC);

  const int waves  = (N + 15) / 16;
  const int gblks  = (waves + 7) / 8;

  // 3) layer 1: transform, init bias, scatter, relu
  gcn_wmma_gemm<HIDC><<<gblks, 256, 0, stream>>>(x, W1t, h1, N);
  long long tot1 = (long long)N * HIDC;
  gcn_init_bias<HIDC><<<(int)((tot1 + 255) / 256), 256, 0, stream>>>(agg1, b1, tot1);
  long long sthr1 = (long long)E * (HIDC / 4);
  gcn_scatter<HIDC><<<(int)((sthr1 + 255) / 256), 256, 0, stream>>>(h1, nrm, row, col, agg1, E);
  gcn_relu<<<(int)((tot1 + 255) / 256), 256, 0, stream>>>(agg1, tot1);

  // 4) layer 2: transform, init bias into d_out, scatter
  gcn_wmma_gemm<OUTC><<<gblks, 256, 0, stream>>>(agg1, W2t, h2, N);
  long long tot2 = (long long)N * OUTC;
  gcn_init_bias<OUTC><<<(int)((tot2 + 255) / 256), 256, 0, stream>>>(out, b2, tot2);
  long long sthr2 = (long long)E * (OUTC / 4);
  gcn_scatter<OUTC><<<(int)((sthr2 + 255) / 256), 256, 0, stream>>>(h2, nrm, row, col, out, E);
}
